// TransformerEncoderWithTemporalAttention_48627619725850
// MI455X (gfx1250) — compile-verified, matrix operand order checked by reading, not run on hardware
//
#include <hip/hip_runtime.h>
#include <hip/hip_bf16.h>

// ---------------------------------------------------------------------------
// Transformer encoder fwd on gfx1250 (MI455X): all matmuls via
// v_wmma_f32_16x16x32_bf16 (wave32 WMMA), fp32 softmax/LN epilogues.
// Attention: block-cooperative double-buffered LDS staging of K/V chunks,
// using async global->LDS copies (ASYNCcnt) when the toolchain exposes them.
// ---------------------------------------------------------------------------

typedef __bf16 bf16_t;
typedef __attribute__((ext_vector_type(16))) __bf16 v16bf;
typedef __attribute__((ext_vector_type(8)))  __bf16 v8bf;
typedef __attribute__((ext_vector_type(8)))  float  v8f;

#define BDIM  4
#define TDIM  2048
#define EDIM  256
#define HDIM  8
#define HD    32
#define QKVN  768     // 3*E
#define FFDIM 1024    // 4*E
#define NROWS (BDIM * TDIM)   // 8192 token rows

union Frag { v16bf v; v8bf p[2]; };

#define WMMA_BF16(A, B, C) \
  __builtin_amdgcn_wmma_f32_16x16x32_bf16(false, (A), false, (B), (short)0, (C), false, false)

// ---- async global->LDS 16B copy (gfx1250 GLOBAL_LOAD_ASYNC_TO_LDS_B128) ----
#if __has_builtin(__builtin_amdgcn_global_load_async_to_lds_b128)
#define HAS_ASYNC_LDS 1
typedef __attribute__((ext_vector_type(4))) int v4i_t;
typedef __attribute__((address_space(1))) v4i_t* gv4i_p;   // global side
typedef __attribute__((address_space(3))) v4i_t* lv4i_p;   // LDS side
#else
#define HAS_ASYNC_LDS 0
#endif

__device__ __forceinline__ void copy16_to_lds(const bf16_t* g, bf16_t* l) {
#if HAS_ASYNC_LDS
  __builtin_amdgcn_global_load_async_to_lds_b128((gv4i_p)g, (lv4i_p)l, 0, 0);
#else
  *(v8bf*)l = *(const v8bf*)g;
#endif
}
__device__ __forceinline__ void async_copy_wait() {
#if HAS_ASYNC_LDS
  asm volatile("s_wait_asynccnt 0" ::: "memory");
#endif
}

// DPP lane permute within a 16-lane row (constant ctrl via template param)
template <int CTRL>
__device__ __forceinline__ float dpp_f(float x) {
  return __int_as_float(
      __builtin_amdgcn_update_dpp(0, __float_as_int(x), CTRL, 0xF, 0xF, true));
}
// max across the 16-lane row group (matches WMMA C/D row striping)
__device__ __forceinline__ float rowmax16(float x) {
  x = fmaxf(x, dpp_f<0xB1>(x));    // quad_perm(1,0,3,2)  : xor 1
  x = fmaxf(x, dpp_f<0x4E>(x));    // quad_perm(2,3,0,1)  : xor 2
  x = fmaxf(x, dpp_f<0x141>(x));   // row_half_mirror     : pairs quads
  x = fmaxf(x, dpp_f<0x140>(x));   // row_mirror          : pairs octs
  return x;
}

// ---------------------------------------------------------------------------
// GEMM: C[M,N] = act(A[M,K](bf16) @ Bt[N,K]^T(bf16) + bias[N])
// Wave computes 32(M) x 64(N): 2 A-frags x 4 B-frags -> 8 WMMAs per 32-wide
// K step. Block = 8 waves = 2(M) x 4(N) -> 64 x 256 block tile.
// ---------------------------------------------------------------------------
template <bool RELU, bool WF32, bool WBF>
__global__ void gemm_wmma(const bf16_t* __restrict__ A,
                          const bf16_t* __restrict__ Bt,
                          const float*  __restrict__ bias,
                          float*        __restrict__ Cf,
                          bf16_t*       __restrict__ Cb,
                          int M, int N, int K) {
  const int wave = threadIdx.x >> 5;
  const int lane = threadIdx.x & 31;
  const int hf   = lane >> 4;     // K-half of the 32-wide chunk
  const int l16  = lane & 15;     // row (A) / col (B) within tile
  const int m0 = blockIdx.y * 64 + (wave >> 2) * 32;
  const int n0 = blockIdx.x * 256 + (wave & 3) * 64;

  const bf16_t* ar0 = A + (size_t)(m0 + l16) * K + hf * 8;
  const bf16_t* ar1 = ar0 + (size_t)16 * K;
  const bf16_t* br0 = Bt + (size_t)(n0 + l16) * K + hf * 8;
  const size_t bstep = (size_t)16 * K;
  const bf16_t* br1 = br0 + bstep;
  const bf16_t* br2 = br0 + 2 * bstep;
  const bf16_t* br3 = br0 + 3 * bstep;

  v8f acc[2][4];
#pragma unroll
  for (int i = 0; i < 2; ++i)
#pragma unroll
    for (int t = 0; t < 4; ++t) acc[i][t] = (v8f){};

  for (int k0 = 0; k0 < K;
       k0 += 32, ar0 += 32, ar1 += 32, br0 += 32, br1 += 32, br2 += 32, br3 += 32) {
    Frag a0, a1, b0, b1, b2, b3;
    a0.p[0] = *(const v8bf*)(ar0);  a0.p[1] = *(const v8bf*)(ar0 + 16);
    a1.p[0] = *(const v8bf*)(ar1);  a1.p[1] = *(const v8bf*)(ar1 + 16);
    b0.p[0] = *(const v8bf*)(br0);  b0.p[1] = *(const v8bf*)(br0 + 16);
    b1.p[0] = *(const v8bf*)(br1);  b1.p[1] = *(const v8bf*)(br1 + 16);
    b2.p[0] = *(const v8bf*)(br2);  b2.p[1] = *(const v8bf*)(br2 + 16);
    b3.p[0] = *(const v8bf*)(br3);  b3.p[1] = *(const v8bf*)(br3 + 16);
    acc[0][0] = WMMA_BF16(a0.v, b0.v, acc[0][0]);
    acc[0][1] = WMMA_BF16(a0.v, b1.v, acc[0][1]);
    acc[0][2] = WMMA_BF16(a0.v, b2.v, acc[0][2]);
    acc[0][3] = WMMA_BF16(a0.v, b3.v, acc[0][3]);
    acc[1][0] = WMMA_BF16(a1.v, b0.v, acc[1][0]);
    acc[1][1] = WMMA_BF16(a1.v, b1.v, acc[1][1]);
    acc[1][2] = WMMA_BF16(a1.v, b2.v, acc[1][2]);
    acc[1][3] = WMMA_BF16(a1.v, b3.v, acc[1][3]);
  }

#pragma unroll
  for (int t = 0; t < 4; ++t) {
    const int col = n0 + t * 16 + l16;
    const float bv = bias ? bias[col] : 0.0f;
#pragma unroll
    for (int i = 0; i < 2; ++i) {
#pragma unroll
      for (int j = 0; j < 8; ++j) {        // D layout: row = hf*8+j, col = l16
        const int row = m0 + i * 16 + hf * 8 + j;
        float v = acc[i][t][j] + bv;
        if (RELU) v = fmaxf(v, 0.0f);
        if (WF32) Cf[(size_t)row * N + col] = v;
        if (WBF)  Cb[(size_t)row * N + col] = (bf16_t)v;
      }
    }
  }
}

// ---------------------------------------------------------------------------
// Online-softmax update for one 16-row tile over a 32-key chunk.
// s0 = even keys, s1 = odd keys. Writes bf16 P (natural key order) to LDS.
// ---------------------------------------------------------------------------
__device__ __forceinline__ void softmax_update(v8f& s0, v8f& s1, float scale,
                                               float* mrow, float* lrow,
                                               v8f& a0, v8f& a1,
                                               unsigned* plds, int hf, int l16) {
#pragma unroll
  for (int j = 0; j < 8; ++j) {
    const float a = s0[j] * scale;          // key kc + 2*l16
    const float c = s1[j] * scale;          // key kc + 2*l16 + 1
    const float mx = rowmax16(fmaxf(a, c));
    const float mnew = fmaxf(mrow[j], mx);
    const float r = __expf(mrow[j] - mnew);
    mrow[j] = mnew;
    lrow[j] *= r;
    a0[j] *= r;
    a1[j] *= r;
    const float p0 = __expf(a - mnew);
    const float p1 = __expf(c - mnew);
    union { bf16_t b; unsigned short u; } u0, u1;
    u0.b = (bf16_t)p0;
    u1.b = (bf16_t)p1;
    plds[(hf * 8 + j) * 16 + l16] = (unsigned)u0.u | ((unsigned)u1.u << 16);
  }
}

// ---------------------------------------------------------------------------
// Flash-style attention. Block = 4 waves working on the SAME (b,h) with four
// consecutive 16-row Q tiles (64 q rows). Each 32-key K/V chunk is staged
// once into double-buffered LDS (cooperative 128x16B copy = 4KB, async when
// available), then all waves build fragments from LDS. One barrier per chunk.
// qkv row layout: [h][{q,k,v}][HD]; vt layout: [b][h][d][t].
// ---------------------------------------------------------------------------
__global__ void __launch_bounds__(128) attention_kernel(
    const bf16_t* __restrict__ qkv,
    const bf16_t* __restrict__ vt,
    bf16_t* __restrict__ out) {
  __shared__ __align__(16) bf16_t kbuf[2][32][32];     // [buf][key][hd]
  __shared__ __align__(16) bf16_t vbuf[2][32][32];     // [buf][hd][t]
  __shared__ __align__(16) unsigned pbuf[4][16 * 16];  // per-wave 16x32 bf16 P

  const int tid  = threadIdx.x;
  const int wave = tid >> 5;
  const int lane = tid & 31;
  const int hf   = lane >> 4;
  const int l16  = lane & 15;

  const int blk = blockIdx.x;            // 0..1023
  const int b   = blk >> 8;              // H*(T/64) = 256 blocks per batch
  const int h   = (blk >> 5) & 7;
  const int q0  = (blk & 31) * 64 + wave * 16;
  const float scale = 0.17677669529663687f;   // 1/sqrt(32)

  // cooperative copy lane mapping: 128 threads x 16B = one 32x32 bf16 tile
  const int crow = tid >> 2;             // 0..31
  const int ccol = (tid & 3) * 8;        // 0,8,16,24
  const bf16_t* ksrc = qkv + (size_t)(b * TDIM + crow) * QKVN + h * 96 + 32 + ccol;
  const bf16_t* vsrc = vt + (size_t)((b * HDIM + h) * HD + crow) * TDIM + ccol;

  // Q fragment (fixed for whole key loop)
  Frag qa;
  {
    const bf16_t* qr = qkv + (size_t)(b * TDIM + q0 + l16) * QKVN + h * 96 + hf * 8;
    qa.p[0] = *(const v8bf*)(qr);
    qa.p[1] = *(const v8bf*)(qr + 16);
  }
  Frag ones;
#pragma unroll
  for (int i = 0; i < 16; ++i) ones.v[i] = (bf16_t)1.0f;

  float mrow[8], lrow[8];
  v8f acc0 = {}, acc1 = {};
#pragma unroll
  for (int j = 0; j < 8; ++j) { mrow[j] = -1e30f; lrow[j] = 0.0f; }

  unsigned* pl = pbuf[wave];
  const bf16_t* plb = (const bf16_t*)pl;

  // preload chunk 0
  copy16_to_lds(ksrc, &kbuf[0][crow][ccol]);
  copy16_to_lds(vsrc, &vbuf[0][crow][ccol]);
  async_copy_wait();
  __syncthreads();

  for (int kc = 0; kc < TDIM; kc += 32) {
    const int cur = (kc >> 5) & 1;
    if (kc + 32 < TDIM) {                 // stream next chunk into other buffer
      copy16_to_lds(ksrc + (size_t)(kc + 32) * QKVN, &kbuf[cur ^ 1][crow][ccol]);
      copy16_to_lds(vsrc + kc + 32, &vbuf[cur ^ 1][crow][ccol]);
      __builtin_prefetch(ksrc + (size_t)(kc + 64) * QKVN, 0, 3);
      __builtin_prefetch(vsrc + kc + 64, 0, 3);
    }

    // ---- K fragments from LDS (tile0 = even keys, tile1 = odd keys) -------
    Frag kb0, kb1;
    kb0.p[0] = *(const v8bf*)&kbuf[cur][2 * l16][hf * 8];
    kb0.p[1] = *(const v8bf*)&kbuf[cur][2 * l16][16 + hf * 8];
    kb1.p[0] = *(const v8bf*)&kbuf[cur][2 * l16 + 1][hf * 8];
    kb1.p[1] = *(const v8bf*)&kbuf[cur][2 * l16 + 1][16 + hf * 8];
    v8f s0 = {}, s1 = {};
    s0 = WMMA_BF16(qa.v, kb0.v, s0);
    s1 = WMMA_BF16(qa.v, kb1.v, s1);

    // ---- V fragments from LDS (natural key order) --------------------------
    Frag vb0, vb1;
    vb0.p[0] = *(const v8bf*)&vbuf[cur][l16][hf * 8];
    vb0.p[1] = *(const v8bf*)&vbuf[cur][l16][16 + hf * 8];
    vb1.p[0] = *(const v8bf*)&vbuf[cur][16 + l16][hf * 8];
    vb1.p[1] = *(const v8bf*)&vbuf[cur][16 + l16][16 + hf * 8];

    softmax_update(s0, s1, scale, mrow, lrow, acc0, acc1, pl, hf, l16);
    asm volatile("s_wait_dscnt 0" ::: "memory");   // wave-local LDS transpose

    Frag pa;
    pa.p[0] = *(const v8bf*)(plb + l16 * 32 + hf * 8);
    pa.p[1] = *(const v8bf*)(plb + l16 * 32 + 16 + hf * 8);

    v8f ls = {};
    ls = WMMA_BF16(pa.v, ones.v, ls);     // row-sums of P
    acc0 = WMMA_BF16(pa.v, vb0.v, acc0);
    acc1 = WMMA_BF16(pa.v, vb1.v, acc1);
#pragma unroll
    for (int j = 0; j < 8; ++j) lrow[j] += ls[j];

    async_copy_wait();   // next-chunk async stores landed in LDS
    __syncthreads();     // all waves done with cur buf; stores visible
  }

#pragma unroll
  for (int j = 0; j < 8; ++j) {
    const int row = b * TDIM + q0 + hf * 8 + j;
    const float inv = 1.0f / lrow[j];
    out[(size_t)row * EDIM + h * HD + l16]      = (bf16_t)(acc0[j] * inv);
    out[(size_t)row * EDIM + h * HD + 16 + l16] = (bf16_t)(acc1[j] * inv);
  }
}

// ---------------------------------------------------------------------------
// x + fourier PE -> f32 + bf16 copies
// ---------------------------------------------------------------------------
__global__ void pe_kernel(const float* __restrict__ x,
                          float* __restrict__ xf, bf16_t* __restrict__ xb) {
  const int idx = blockIdx.x * 256 + threadIdx.x;   // B*T*E exactly
  const int e = idx & 255;
  const int t = (idx >> 8) & 2047;
  const int j = e & 127;
  const float f = __powf(10000.0f, -(2.0f * (float)j) / 256.0f);
  const float s = (float)t * f;
  const float pe = (e < 128) ? __sinf(s) : __cosf(s);
  const float v = x[idx] + pe;
  xf[idx] = v;
  xb[idx] = (bf16_t)v;
}

// ---------------------------------------------------------------------------
// y = LN(resid + delta) ; one 256-wide row per block
// ---------------------------------------------------------------------------
__global__ void add_ln_kernel(const float* __restrict__ resid,
                              const float* __restrict__ delta,
                              const float* __restrict__ g,
                              const float* __restrict__ bb,
                              float* __restrict__ of, bf16_t* __restrict__ ob) {
  __shared__ float red[8];
  const int row = blockIdx.x;
  const int e = threadIdx.x;
  const float v = resid[(size_t)row * EDIM + e] + delta[(size_t)row * EDIM + e];

  float s = v;
  for (int m = 1; m < 32; m <<= 1) s += __shfl_xor(s, m, 32);
  if ((threadIdx.x & 31) == 0) red[threadIdx.x >> 5] = s;
  __syncthreads();
  float tot = 0.0f;
#pragma unroll
  for (int i = 0; i < 8; ++i) tot += red[i];
  const float mu = tot * (1.0f / EDIM);
  const float d = v - mu;

  float s2 = d * d;
  for (int m = 1; m < 32; m <<= 1) s2 += __shfl_xor(s2, m, 32);
  __syncthreads();
  if ((threadIdx.x & 31) == 0) red[threadIdx.x >> 5] = s2;
  __syncthreads();
  float tot2 = 0.0f;
#pragma unroll
  for (int i = 0; i < 8; ++i) tot2 += red[i];
  const float var = tot2 * (1.0f / EDIM);

  const float y = d * rsqrtf(var + 1e-5f) * g[e] + bb[e];
  of[(size_t)row * EDIM + e] = y;
  ob[(size_t)row * EDIM + e] = (bf16_t)y;
}

// ---------------------------------------------------------------------------
// fp32 (R x C) -> bf16 transposed (C x R)  [weight prep]
// ---------------------------------------------------------------------------
__global__ void tcast_kernel(const float* __restrict__ src,
                             bf16_t* __restrict__ dst, int R, int C) {
  const int idx = blockIdx.x * 256 + threadIdx.x;
  if (idx >= R * C) return;
  const int r = idx / C, c = idx % C;
  dst[(size_t)c * R + r] = (bf16_t)src[(size_t)r * C + c];
}

// ---------------------------------------------------------------------------
// Extract V from qkv and transpose to [b][h][d][t] via LDS 32x32 tiles
// grid: (T/32, B*H), block: (32, 8)
// ---------------------------------------------------------------------------
__global__ void vtrans_kernel(const bf16_t* __restrict__ qkv,
                              bf16_t* __restrict__ vt) {
  __shared__ bf16_t tile[32][33];
  const int bh = blockIdx.y;            // b*8 + h
  const int b = bh >> 3, h = bh & 7;
  const int t0 = blockIdx.x * 32;
#pragma unroll
  for (int i = 0; i < 32; i += 8) {     // rows = t, cols = d (coalesced in d)
    const int t = t0 + threadIdx.y + i;
    tile[threadIdx.y + i][threadIdx.x] =
        qkv[(size_t)(b * TDIM + t) * QKVN + h * 96 + 64 + threadIdx.x];
  }
  __syncthreads();
#pragma unroll
  for (int i = 0; i < 32; i += 8) {     // rows = d, cols = t (coalesced in t)
    const int d = threadIdx.y + i;
    vt[(size_t)(bh * 32 + d) * TDIM + t0 + threadIdx.x] = tile[threadIdx.x][d];
  }
}

// ---------------------------------------------------------------------------
extern "C" void kernel_launch(void* const* d_in, const int* in_sizes, int n_in,
                              void* d_out, int out_size, void* d_ws, size_t ws_size,
                              hipStream_t stream) {
  (void)in_sizes; (void)n_in; (void)out_size; (void)ws_size;
  const float* x_in  = (const float*)d_in[0];
  const float* qkv_w = (const float*)d_in[1];
  const float* qkv_b = (const float*)d_in[2];
  const float* out_w = (const float*)d_in[3];
  const float* out_b = (const float*)d_in[4];
  const float* ff1_w = (const float*)d_in[5];
  const float* ff1_b = (const float*)d_in[6];
  const float* ff2_w = (const float*)d_in[7];
  const float* ff2_b = (const float*)d_in[8];
  const float* ln1_g = (const float*)d_in[9];
  const float* ln1_b = (const float*)d_in[10];
  const float* ln2_g = (const float*)d_in[11];
  const float* ln2_b = (const float*)d_in[12];
  float* out = (float*)d_out;

  char* ws = (char*)d_ws;
  size_t off = 0;
  auto alloc = [&](size_t bytes) -> void* {
    void* p = ws + off;
    off += (bytes + 255) & ~(size_t)255;
    return p;
  };
  float*  x_f32   = (float*)alloc((size_t)NROWS * EDIM * 4);
  bf16_t* x_bf    = (bf16_t*)alloc((size_t)NROWS * EDIM * 2);
  bf16_t* qkv_bf  = (bf16_t*)alloc((size_t)NROWS * QKVN * 2);
  bf16_t* vt      = (bf16_t*)alloc((size_t)BDIM * HDIM * HD * TDIM * 2);
  bf16_t* attn_bf = (bf16_t*)alloc((size_t)NROWS * EDIM * 2);
  float*  tmp_f32 = (float*)alloc((size_t)NROWS * EDIM * 4);
  bf16_t* h_bf    = (bf16_t*)alloc((size_t)NROWS * FFDIM * 2);
  bf16_t* qkvw_t  = (bf16_t*)alloc((size_t)4 * EDIM * QKVN * 2);
  bf16_t* outw_t  = (bf16_t*)alloc((size_t)4 * EDIM * EDIM * 2);
  bf16_t* ff1w_t  = (bf16_t*)alloc((size_t)4 * EDIM * FFDIM * 2);
  bf16_t* ff2w_t  = (bf16_t*)alloc((size_t)4 * FFDIM * EDIM * 2);

  // Weight cast + transpose (bf16, N-major) for all layers
  for (int l = 0; l < 4; ++l) {
    tcast_kernel<<<(EDIM * QKVN + 255) / 256, 256, 0, stream>>>(
        qkv_w + (size_t)l * EDIM * QKVN, qkvw_t + (size_t)l * EDIM * QKVN, EDIM, QKVN);
    tcast_kernel<<<(EDIM * EDIM + 255) / 256, 256, 0, stream>>>(
        out_w + (size_t)l * EDIM * EDIM, outw_t + (size_t)l * EDIM * EDIM, EDIM, EDIM);
    tcast_kernel<<<(EDIM * FFDIM + 255) / 256, 256, 0, stream>>>(
        ff1_w + (size_t)l * EDIM * FFDIM, ff1w_t + (size_t)l * EDIM * FFDIM, EDIM, FFDIM);
    tcast_kernel<<<(FFDIM * EDIM + 255) / 256, 256, 0, stream>>>(
        ff2_w + (size_t)l * FFDIM * EDIM, ff2w_t + (size_t)l * FFDIM * EDIM, FFDIM, EDIM);
  }

  pe_kernel<<<(NROWS * EDIM) / 256, 256, 0, stream>>>(x_in, x_f32, x_bf);

  const dim3 gq(QKVN / 256, NROWS / 64);    // (3, 128)
  const dim3 go(EDIM / 256, NROWS / 64);    // (1, 128)
  const dim3 g1(FFDIM / 256, NROWS / 64);   // (4, 128)
  const dim3 g2(EDIM / 256, NROWS / 64);    // (1, 128)
  const dim3 vtb(32, 8);
  const dim3 vtg(TDIM / 32, BDIM * HDIM);
  const int attn_blocks = BDIM * HDIM * (TDIM / 64);   // 1024 blocks x 128 thr

  for (int l = 0; l < 4; ++l) {
    // QKV projection (bf16 out, consumed by attention)
    gemm_wmma<false, false, true><<<gq, 256, 0, stream>>>(
        x_bf, qkvw_t + (size_t)l * EDIM * QKVN, qkv_b + (size_t)l * QKVN,
        nullptr, qkv_bf, NROWS, QKVN, EDIM);

    vtrans_kernel<<<vtg, vtb, 0, stream>>>(qkv_bf, vt);

    attention_kernel<<<attn_blocks, 128, 0, stream>>>(qkv_bf, vt, attn_bf);

    // output projection (f32 out for residual)
    gemm_wmma<false, true, false><<<go, 256, 0, stream>>>(
        attn_bf, outw_t + (size_t)l * EDIM * EDIM, out_b + (size_t)l * EDIM,
        tmp_f32, nullptr, NROWS, EDIM, EDIM);

    add_ln_kernel<<<NROWS, 256, 0, stream>>>(
        x_f32, tmp_f32, ln1_g + (size_t)l * EDIM, ln1_b + (size_t)l * EDIM,
        x_f32, x_bf);

    // FF1 + ReLU (bf16 out)
    gemm_wmma<true, false, true><<<g1, 256, 0, stream>>>(
        x_bf, ff1w_t + (size_t)l * EDIM * FFDIM, ff1_b + (size_t)l * FFDIM,
        nullptr, h_bf, NROWS, FFDIM, EDIM);

    // FF2 (f32 out)
    gemm_wmma<false, true, false><<<g2, 256, 0, stream>>>(
        h_bf, ff2w_t + (size_t)l * FFDIM * EDIM, ff2_b + (size_t)l * EDIM,
        tmp_f32, nullptr, NROWS, EDIM, FFDIM);

    float* lnout = (l == 3) ? out : x_f32;
    add_ln_kernel<<<NROWS, 256, 0, stream>>>(
        x_f32, tmp_f32, ln2_g + (size_t)l * EDIM, ln2_b + (size_t)l * EDIM,
        lnout, x_bf);
  }
}
